// Cascade_CNN_RNN_64596308132089
// MI455X (gfx1250) — matrix operand order repaired and checked by reading, not compiled, not run on hardware
//
#include <hip/hip_runtime.h>
#include <hip/hip_bf16.h>

typedef __attribute__((ext_vector_type(16))) _Float16 v16h;
typedef __attribute__((ext_vector_type(8)))  _Float16 v8h;
typedef __attribute__((ext_vector_type(4)))  _Float16 v4h;
typedef __attribute__((ext_vector_type(8)))  float    v8f;

#define DEV __device__ __forceinline__

DEV float clip01(float v) { return fminf(fmaxf(v, 0.f), 1.f); }
DEV float clip11(float v) { return fminf(fmaxf(v, -1.f), 1.f); }

// ---- A fragment (16x32 f16, wave32) via two b128 LDS loads per lane ----
// ISA 7.12.2: lane l, elements 0..7  -> row (l&15), K = (l>>4)*8 + 0..7
//             elements 8..15        -> row (l&15), K = 16 + (l>>4)*8 + 0..7
DEV v16h load_a_frag(const _Float16* __restrict__ A, int lda, int lane) {
  const int r  = lane & 15;
  const int kh = (lane >> 4) << 3;
  const v8h lo = *(const v8h*)(A + r * lda + kh);
  const v8h hi = *(const v8h*)(A + r * lda + 16 + kh);
  v16h a;
#pragma unroll
  for (int i = 0; i < 8; ++i) { a[i] = lo[i]; a[i + 8] = hi[i]; }
  return a;
}

DEV v8f wmma_f16(v16h a, v16h b, v8f c) {
  return __builtin_amdgcn_wmma_f32_16x16x32_f16(false, a, false, b, (short)0, c,
                                                false, false);
}

// =====================================================================
// Weight pre-swizzle: W [N][K] fp32 row-major -> fragment-linear f16.
// Chunk (kt,nt) = 512 halfs; chunk[l*16 + i] = W[nt*16 + (l&15)]
//                    [kt*32 + (i>>3)*16 + (l>>4)*8 + (i&7)]
// so a wave's B fragment is one contiguous 32B load per lane.
// =====================================================================
__global__ void swizzle_b_kernel(const float* __restrict__ W,
                                 _Float16* __restrict__ out, int N, int K) {
  const int idx = blockIdx.x * 256 + threadIdx.x;
  if (idx >= N * K) return;
  const int NT = N >> 4;
  const int i  = idx & 15;
  const int l  = (idx >> 4) & 31;
  const int cn = idx >> 9;
  const int nt = cn % NT;
  const int kt = cn / NT;
  const int k  = kt * 32 + ((i >> 3) << 4) + ((l >> 4) << 3) + (i & 7);
  const int n  = nt * 16 + (l & 15);
  out[idx] = (_Float16)W[(size_t)n * K + k];
}

// =====================================================================
// Fused quantize + conv1(1->16) + conv2(16->32), one image per block.
// =====================================================================
__global__ __launch_bounds__(128) void conv_kernel(
    const float* __restrict__ x,   // [B,10,10,11]
    const float* __restrict__ w1,  // [16,1,3,3]
    const float* __restrict__ w2,  // [32,16,3,3]
    _Float16* __restrict__ a2,     // [B,3520]
    int t) {
  __shared__ alignas(16) float xi[110 + 2];
  __shared__ alignas(16) float a1[16 * 110];
  __shared__ alignas(16) float cw1[144];
  __shared__ alignas(16) float cw2[4608];

  const int b = blockIdx.x;
  const int tid = threadIdx.x;

  for (int i = tid; i < 36; i += 128)   ((float4*)cw1)[i] = ((const float4*)w1)[i];
  for (int i = tid; i < 1152; i += 128) ((float4*)cw2)[i] = ((const float4*)w2)[i];
  if (tid < 110) {
    const int h = tid / 11, w = tid - h * 11;
    const float v = clip11(x[((b * 10 + t) * 10 + h) * 11 + w]);
    xi[tid] = rintf(v * 128.f) * (1.f / 128.f);  // bipolar SC quantize
  }
  __syncthreads();

  const int h = tid / 11, w = tid - h * 11;
  if (tid < 110) {
#pragma unroll
    for (int oc = 0; oc < 16; ++oc) {
      float acc = 0.f;
#pragma unroll
      for (int kh = 0; kh < 3; ++kh) {
        const int hh = h + kh - 1;
        if ((unsigned)hh >= 10u) continue;
#pragma unroll
        for (int kw = 0; kw < 3; ++kw) {
          const int ww = w + kw - 1;
          if ((unsigned)ww >= 11u) continue;
          acc = fmaf(xi[hh * 11 + ww], cw1[oc * 9 + kh * 3 + kw], acc);
        }
      }
      a1[oc * 110 + tid] = clip01(acc);
    }
  }
  __syncthreads();

  if (tid < 110) {
    for (int oc = 0; oc < 32; ++oc) {
      float acc = 0.f;
      for (int ic = 0; ic < 16; ++ic) {
#pragma unroll
        for (int kh = 0; kh < 3; ++kh) {
          const int hh = h + kh - 1;
          if ((unsigned)hh >= 10u) continue;
#pragma unroll
          for (int kw = 0; kw < 3; ++kw) {
            const int ww = w + kw - 1;
            if ((unsigned)ww >= 11u) continue;
            acc = fmaf(a1[ic * 110 + hh * 11 + ww],
                       cw2[((oc * 16 + ic) * 3 + kh) * 3 + kw], acc);
          }
        }
      }
      a2[(size_t)b * 3520 + oc * 110 + tid] = (_Float16)clip01(acc);
    }
  }
}

// =====================================================================
// fc3: a3 = clip01(a2[B,3520] @ W.T)  (W fragment-linear, N=256, K=3520)
// 64x64 block tile, 4 waves. Double-buffered LDS A staging; B fragments
// loaded straight from the fragment-linear global buffer (L2-resident).
// Issue order per k-step: B frags -> next-A prefetch -> WMMA (waits only
// for B; prefetch stays outstanding, in-order completion) -> LDS store.
// =====================================================================
__global__ __launch_bounds__(128) void fc3_kernel(
    const _Float16* __restrict__ a2,     // [B,3520]
    const _Float16* __restrict__ wfrag,  // frag-linear [110 kt][16 nt][512]
    _Float16* __restrict__ a3) {         // [B,256]
  __shared__ alignas(16) _Float16 As[2][64 * 32];

  const int tid = threadIdx.x, lane = tid & 31, wv = tid >> 5;
  const int m0 = blockIdx.x * 64, n0 = blockIdx.y * 64;

  // this thread's two 8-half staging chunks: rows r0 and r0+32, k-offset k0
  const int r0 = tid >> 2, k0 = (tid & 3) * 8;
  const _Float16* arow0 = a2 + (size_t)(m0 + r0) * 3520 + k0;
  const _Float16* arow1 = a2 + (size_t)(m0 + r0 + 32) * 3520 + k0;

  // prologue: stage k-tile 0
  *(v8h*)&As[0][r0 * 32 + k0]        = *(const v8h*)(arow0);
  *(v8h*)&As[0][(r0 + 32) * 32 + k0] = *(const v8h*)(arow1);
  __syncthreads();

  v8f acc[4];
  const v8f z = {};
#pragma unroll
  for (int j = 0; j < 4; ++j) acc[j] = z;

  for (int kt = 0; kt < 110; ++kt) {
    const int cur = kt & 1, nxt = cur ^ 1;

    // 1) B fragments for this k-step (consumed by the wmma chain)
    const _Float16* bb =
        wfrag + ((size_t)kt * 16 + (n0 >> 4)) * 512 + lane * 16;
    const v16h b0 = *(const v16h*)(bb);
    const v16h b1 = *(const v16h*)(bb + 512);
    const v16h b2 = *(const v16h*)(bb + 1024);
    const v16h b3 = *(const v16h*)(bb + 1536);

    // 2) prefetch next A tile into registers (outstanding across WMMAs)
    v8h p0 = {}, p1 = {};
    if (kt + 1 < 110) {
      p0 = *(const v8h*)(arow0 + (size_t)(kt + 1) * 32);
      p1 = *(const v8h*)(arow1 + (size_t)(kt + 1) * 32);
    }

    // 3) compute
    const v16h a = load_a_frag(&As[cur][wv * 16 * 32], 32, lane);
    acc[0] = wmma_f16(a, b0, acc[0]);
    acc[1] = wmma_f16(a, b1, acc[1]);
    acc[2] = wmma_f16(a, b2, acc[2]);
    acc[3] = wmma_f16(a, b3, acc[3]);

    // 4) commit prefetched tile to the other LDS buffer
    if (kt + 1 < 110) {
      *(v8h*)&As[nxt][r0 * 32 + k0]        = p0;
      *(v8h*)&As[nxt][(r0 + 32) * 32 + k0] = p1;
    }
    __syncthreads();
  }

  const int half = lane >> 4, nl = lane & 15;
#pragma unroll
  for (int j = 0; j < 4; ++j)
#pragma unroll
    for (int v = 0; v < 8; ++v) {
      const int m = m0 + wv * 16 + v + 8 * half;  // D: M = vgpr + 8*(lane/16)
      const int n = n0 + j * 16 + nl;
      a3[(size_t)m * 256 + n] = (_Float16)clip01(acc[j][v]);
    }
}

// =====================================================================
// Fused MGU cell. Wave owns 16 rows x N=128 -> gate cols (c, c+64) are
// accumulators j and j+4 of the SAME lane; gate math is register-only.
// B-fragment loads grouped in fours so waits amortize.
// =====================================================================
__global__ __launch_bounds__(128) void mgu_kernel(
    const _Float16* __restrict__ a3,    // [B,256]
    const _Float16* __restrict__ wihF,  // frag-linear [8 kt][8 nt][512]
    const _Float16* __restrict__ whhF,  // frag-linear [2 kt][8 nt][512]
    float* __restrict__ hx) {           // [B,64] in/out
  __shared__ alignas(16) _Float16 hqh[64 * 64];
  __shared__ alignas(16) _Float16 As[64 * 32];

  const int tid = threadIdx.x, lane = tid & 31, wv = tid >> 5;
  const int m0 = blockIdx.x * 64;

  // stage quantized hidden state (multiples of 1/128: exact in f16)
  for (int i = tid; i < 1024; i += 128) {
    const int r = i >> 4, c4 = (i & 15) * 4;
    const float4 v = *(const float4*)&hx[(size_t)(m0 + r) * 64 + c4];
    v4h q;
    q[0] = (_Float16)(rintf(clip11(v.x) * 128.f) * (1.f / 128.f));
    q[1] = (_Float16)(rintf(clip11(v.y) * 128.f) * (1.f / 128.f));
    q[2] = (_Float16)(rintf(clip11(v.z) * 128.f) * (1.f / 128.f));
    q[3] = (_Float16)(rintf(clip11(v.w) * 128.f) * (1.f / 128.f));
    *(v4h*)&hqh[r * 64 + c4] = q;
  }
  __syncthreads();

  const v8f z = {};
  v8f gh[8], gi[8];
#pragma unroll
  for (int j = 0; j < 8; ++j) { gh[j] = z; gi[j] = z; }

  // gh = hq @ whh.T  (K=64; A in LDS, B direct from global frag buffer)
#pragma unroll
  for (int kt = 0; kt < 2; ++kt) {
    const v16h a = load_a_frag(&hqh[wv * 16 * 64 + kt * 32], 64, lane);
    const _Float16* bb = whhF + (size_t)kt * 8 * 512 + lane * 16;
#pragma unroll
    for (int g = 0; g < 2; ++g) {
      v16h bf[4];
#pragma unroll
      for (int j = 0; j < 4; ++j)
        bf[j] = *(const v16h*)(bb + (g * 4 + j) * 512);
#pragma unroll
      for (int j = 0; j < 4; ++j)
        gh[g * 4 + j] = wmma_f16(a, bf[j], gh[g * 4 + j]);
    }
  }

  // gi = a3 @ wih.T  (K=256; A staged per k-tile)
  for (int kt = 0; kt < 8; ++kt) {
    __syncthreads();
    for (int i = tid; i < 256; i += 128) {
      const int r = i >> 2, ko = (i & 3) * 8;
      *(v8h*)&As[r * 32 + ko] =
          *(const v8h*)&a3[(size_t)(m0 + r) * 256 + kt * 32 + ko];
    }
    __syncthreads();
    const v16h a = load_a_frag(&As[wv * 16 * 32], 32, lane);
    const _Float16* bb = wihF + (size_t)kt * 8 * 512 + lane * 16;
#pragma unroll
    for (int g = 0; g < 2; ++g) {
      v16h bf[4];
#pragma unroll
      for (int j = 0; j < 4; ++j)
        bf[j] = *(const v16h*)(bb + (g * 4 + j) * 512);
#pragma unroll
      for (int j = 0; j < 4; ++j)
        gi[g * 4 + j] = wmma_f16(a, bf[j], gi[g * 4 + j]);
    }
  }

  // gate math, all in registers
  const int half = lane >> 4, nl = lane & 15;
#pragma unroll
  for (int j = 0; j < 4; ++j)
#pragma unroll
    for (int v = 0; v < 8; ++v) {
      const int mrow = wv * 16 + v + 8 * half;
      const int c = j * 16 + nl;  // 0..63
      const float gif = gi[j][v], gin = gi[j + 4][v];
      const float ghf = gh[j][v], ghn = gh[j + 4][v];
      const float hq = (float)hqh[mrow * 64 + c];
      const float f  = clip01(0.5f * (gif + ghf) + 0.5f);  // hard sigmoid
      const float ng = clip11(gin + f * ghn);               // hard tanh
      hx[(size_t)(m0 + mrow) * 64 + c] = (1.f - f) * ng + f * hq;
    }
}

// ---- prep / epilogue ----
__global__ void zero_kernel(float* p, int n) {
  const int i = blockIdx.x * 256 + threadIdx.x;
  if (i < n) p[i] = 0.f;
}

__global__ void fc5_kernel(const float* __restrict__ hx,
                           const float* __restrict__ w,  // [7,64]
                           float* __restrict__ out, int B) {
  const int i = blockIdx.x * 256 + threadIdx.x;
  if (i < B * 7) {
    const int b = i / 7, c = i - b * 7;
    float acc = 0.f;
#pragma unroll
    for (int k = 0; k < 64; ++k)
      acc = fmaf(hx[(size_t)b * 64 + k], w[c * 64 + k], acc);
    out[i] = acc;
  }
}

extern "C" void kernel_launch(void* const* d_in, const int* in_sizes, int n_in,
                              void* d_out, int out_size, void* d_ws, size_t ws_size,
                              hipStream_t stream) {
  (void)n_in; (void)out_size; (void)ws_size;
  const float* x    = (const float*)d_in[0];  // [B,10,10,11]
  const float* w1   = (const float*)d_in[1];  // [16,1,3,3]
  const float* w2   = (const float*)d_in[2];  // [32,16,3,3]
  const float* fc3w = (const float*)d_in[3];  // [256,3520]
  const float* wih  = (const float*)d_in[4];  // [128,256]
  const float* whh  = (const float*)d_in[5];  // [128,64]
  const float* fc5w = (const float*)d_in[6];  // [7,64]
  const int B = in_sizes[0] / (10 * 10 * 11); // 2048

  // workspace layout (256B-aligned slices)
  char* ws = (char*)d_ws;
  size_t off = 0;
  auto take = [&](size_t bytes) -> char* {
    char* p = ws + off;
    off = (off + bytes + 255) & ~(size_t)255;
    return p;
  };
  _Float16* a2   = (_Float16*)take((size_t)B * 3520 * 2);
  _Float16* a3   = (_Float16*)take((size_t)B * 256 * 2);
  float*    hx   = (float*)   take((size_t)B * 64 * 4);
  _Float16* fc3F = (_Float16*)take((size_t)3520 * 256 * 2);
  _Float16* wihF = (_Float16*)take((size_t)256 * 128 * 2);
  _Float16* whhF = (_Float16*)take((size_t)64 * 128 * 2);

  // one-time weight fragment-swizzle + hidden-state init
  swizzle_b_kernel<<<(256 * 3520 + 255) / 256, 256, 0, stream>>>(fc3w, fc3F, 256, 3520);
  swizzle_b_kernel<<<(128 * 256 + 255) / 256, 256, 0, stream>>>(wih, wihF, 128, 256);
  swizzle_b_kernel<<<(128 * 64 + 255) / 256, 256, 0, stream>>>(whh, whhF, 128, 64);
  zero_kernel<<<(B * 64 + 255) / 256, 256, 0, stream>>>(hx, B * 64);

  // 10-step scan
  for (int t = 0; t < 10; ++t) {
    conv_kernel<<<B, 128, 0, stream>>>(x, w1, w2, a2, t);
    fc3_kernel<<<dim3(B / 64, 256 / 64), 128, 0, stream>>>(a2, fc3F, a3);
    mgu_kernel<<<B / 64, 128, 0, stream>>>(a3, wihF, whhF, hx);
  }

  fc5_kernel<<<(B * 7 + 255) / 256, 256, 0, stream>>>(hx, fc5w, (float*)d_out, B);
}
